// mamba_ref_1030792151360
// MI455X (gfx1250) — matrix operand order repaired and checked by reading, not compiled, not run on hardware
//
#include <hip/hip_runtime.h>
#include <math.h>

// ---- problem constants (match reference) ----
#define D_MODEL 2048
#define D_STATE 16
#define D_CONV  4
#define D_INNER 4096
#define DT_RANK 128
#define BATCH   4
#define SEQLEN  2048
#define ML      (BATCH * SEQLEN)          // 8192 GEMM rows
#define XZ_N    (2 * D_INNER)             // 8192
#define XD_N    (DT_RANK + 2 * D_STATE)   // 160

// d_out flat layout: out | new_conv_state | last_state
#define OUT_MAIN ((size_t)ML * D_MODEL)
#define CS_OFF   OUT_MAIN
#define SS_OFF   (OUT_MAIN + (size_t)BATCH * D_INNER * D_CONV)

typedef __attribute__((ext_vector_type(16))) __bf16 v16bf;
typedef __attribute__((ext_vector_type(8)))  __bf16 v8bf;
typedef __attribute__((ext_vector_type(8)))  float  v8f;

__device__ __forceinline__ float sigmoidf_(float x) { return 1.0f / (1.0f + __expf(-x)); }

// ---------------------------------------------------------------------------
// f32 -> bf16 conversion (one-time, trivial at 23.3 TB/s)
// ---------------------------------------------------------------------------
__global__ void f32_to_bf16_kernel(const float* __restrict__ src,
                                   __bf16* __restrict__ dst, long n) {
  long i = (long)blockIdx.x * blockDim.x + threadIdx.x;
  if (i < n) dst[i] = (__bf16)src[i];
}

// ---------------------------------------------------------------------------
// C[M,N] = A[M,K](bf16) @ W[N,K](bf16)^T, f32 accumulate.
// One wave per 16 x (NT*16) output strip. NT is compile-time, grids are exact
// multiples -> straight-line kernel, no EXEC masking anywhere (WMMA-safe).
//
// Fragment layouts used here (contiguous k, same for A and B):
//   A 16x32 bf16 : lane half 0 -> K = k0+0..7 (regs0-3), k0+8..15 (regs4-7)
//                  lane half 1 -> K = k0+16..23,          k0+24..31
//   B 32x16 bf16 : lane = column n; half 0 -> K = k0+0..15, half 1 -> +16..31
//   C/D 16x16 f32: reg r -> (M = r + 8*half, N = lane&15)
// ---------------------------------------------------------------------------
template <int NT>
__global__ void __launch_bounds__(256)
gemm_bf16_wmma(const __bf16* __restrict__ A, int lda,
               const __bf16* __restrict__ W, int ldw,
               float* __restrict__ C, int ldc,
               int M, int K) {
  const int lane = threadIdx.x & 31;
  const int wave = (int)((blockIdx.x * blockDim.x + threadIdx.x) >> 5);
  const int tilesM = M >> 4;
  const int tm = wave % tilesM;
  const int tg = wave / tilesM;

  const int half = lane >> 4;
  const int l15  = lane & 15;

  const __bf16* arow  = A + (size_t)(tm * 16 + l15) * lda + half * 16;
  const __bf16* wbase = W + (size_t)(tg * NT * 16 + l15) * ldw + half * 16;

  v8f acc[NT];
#pragma unroll
  for (int t = 0; t < NT; ++t) acc[t] = (v8f){};

  for (int k0 = 0; k0 < K; k0 += 32) {
    union { v16bf v; v8bf h[2]; } a;
    a.h[0] = *(const v8bf*)(arow + k0);        // K = k0 + half*16 + 0..7
    a.h[1] = *(const v8bf*)(arow + k0 + 8);    // K = k0 + half*16 + 8..15
#pragma unroll
    for (int t = 0; t < NT; ++t) {
      v16bf b = *(const v16bf*)(wbase + (size_t)t * 16 * ldw + k0);
      acc[t] = __builtin_amdgcn_wmma_f32_16x16x32_bf16(
          false, a.v, false, b, (short)0, acc[t], false, false);
    }
  }

#pragma unroll
  for (int t = 0; t < NT; ++t) {
    float* crow = C + (size_t)(tm * 16 + half * 8) * ldc + (tg * NT + t) * 16 + l15;
#pragma unroll
    for (int r = 0; r < 8; ++r) crow[(size_t)r * ldc] = acc[t][r];
  }
}

// ---------------------------------------------------------------------------
// Depthwise causal conv (4 taps) + bias + SiLU -> u (bf16)
// ---------------------------------------------------------------------------
__global__ void conv_silu_kernel(const float* __restrict__ xz,
                                 const float* __restrict__ cw,
                                 const float* __restrict__ cb,
                                 __bf16* __restrict__ u_bf) {
  long idx = (long)blockIdx.x * blockDim.x + threadIdx.x;
  if (idx >= (long)ML * D_INNER) return;
  int  d    = (int)(idx & (D_INNER - 1));
  long bl   = idx >> 12;
  int  l    = (int)(bl & (SEQLEN - 1));
  long brow = bl - l;
  float acc = cb[d];
#pragma unroll
  for (int j = 0; j < D_CONV; ++j) {
    int ll = l - (D_CONV - 1) + j;
    if (ll >= 0)
      acc += cw[d * D_CONV + j] * xz[(brow + ll) * (long)XZ_N + d];
  }
  float uv = acc * sigmoidf_(acc);
  u_bf[idx] = (__bf16)uv;
}

// new_conv_state[b][d][j] = x[b][L-4+j][d]  (pre-activation x)
__global__ void conv_state_kernel(const float* __restrict__ xz,
                                  float* __restrict__ out_cs) {
  int idx = blockIdx.x * blockDim.x + threadIdx.x;
  if (idx >= BATCH * D_INNER * D_CONV) return;
  int j = idx & (D_CONV - 1);
  int d = (idx >> 2) & (D_INNER - 1);
  int b = idx >> 14;
  int l = SEQLEN - D_CONV + j;
  out_cs[idx] = xz[((long)(b * SEQLEN + l)) * XZ_N + d];
}

// ---------------------------------------------------------------------------
// Selective scan: one thread per (b,d); 16-wide state in registers.
// Fuses softplus(dt+bias), dBu, y = (ys + u*D) * silu(z). dt in xz[:,0:4096),
// z in xz[:,4096:). Writes y (bf16) and final ssm state.
// ---------------------------------------------------------------------------
__global__ void __launch_bounds__(256)
scan_kernel(const float* __restrict__ xz,
            const __bf16* __restrict__ u_bf,
            const float* __restrict__ xdbl,   // [ML,160]: dt|B|C
            const float* __restrict__ dtb,
            const float* __restrict__ Amat,   // [16, D_INNER]
            const float* __restrict__ Dvec,
            __bf16* __restrict__ y_bf,
            float* __restrict__ out_ss) {
  int idx = blockIdx.x * blockDim.x + threadIdx.x;
  if (idx >= BATCH * D_INNER) return;
  int d = idx & (D_INNER - 1);
  int b = idx >> 12;

  float Ad[D_STATE], s[D_STATE];
#pragma unroll
  for (int n = 0; n < D_STATE; ++n) { Ad[n] = Amat[n * D_INNER + d]; s[n] = 0.0f; }
  const float bias = dtb[d];
  const float Dd   = Dvec[d];

  for (int l = 0; l < SEQLEN; ++l) {
    long  row   = (long)(b * SEQLEN + l);
    float dt    = xz[row * XZ_N + d];
    float z     = xz[row * XZ_N + D_INNER + d];
    float uv    = (float)u_bf[row * D_INNER + d];
    float x     = dt + bias;
    float delta = (x > 20.0f) ? x : log1pf(__expf(x));   // softplus
    float dbu   = delta * uv;

    const float4* Bp = (const float4*)(xdbl + row * XD_N + DT_RANK);
    const float4* Cp = (const float4*)(xdbl + row * XD_N + DT_RANK + D_STATE);
    float y = 0.0f;
#pragma unroll
    for (int q = 0; q < 4; ++q) {
      float4 Bv = Bp[q], Cv = Cp[q];
      float Ba[4] = {Bv.x, Bv.y, Bv.z, Bv.w};
      float Ca[4] = {Cv.x, Cv.y, Cv.z, Cv.w};
#pragma unroll
      for (int r = 0; r < 4; ++r) {
        int n = q * 4 + r;
        float dA = __expf(delta * Ad[n]);
        s[n] = dA * s[n] + Ba[r] * dbu;
        y += s[n] * Ca[r];
      }
    }
    float yo = (y + uv * Dd) * (z * sigmoidf_(z));
    y_bf[row * D_INNER + d] = (__bf16)yo;
  }
#pragma unroll
  for (int n = 0; n < D_STATE; ++n)
    out_ss[((long)b * D_STATE + n) * D_INNER + d] = s[n];
}

// ---------------------------------------------------------------------------
extern "C" void kernel_launch(void* const* d_in, const int* in_sizes, int n_in,
                              void* d_out, int out_size, void* d_ws, size_t ws_size,
                              hipStream_t stream) {
  (void)in_sizes; (void)n_in; (void)out_size; (void)ws_size;
  const float* hs   = (const float*)d_in[0];
  const float* ipw  = (const float*)d_in[6];
  const float* cw   = (const float*)d_in[7];
  const float* cb   = (const float*)d_in[8];
  const float* xpw  = (const float*)d_in[9];
  const float* dpw  = (const float*)d_in[10];
  const float* dpb  = (const float*)d_in[11];
  const float* Amat = (const float*)d_in[12];
  const float* Dvec = (const float*)d_in[13];
  const float* opw  = (const float*)d_in[14];
  float* out = (float*)d_out;

  char* ws = (char*)d_ws;
  size_t off = 0;
  auto carve = [&](size_t bytes) -> void* {
    void* p = ws + off;
    off = (off + bytes + 255) & ~(size_t)255;
    return p;
  };
  float*  xz      = (float*) carve((size_t)ML * XZ_N * 4);
  float*  xdbl    = (float*) carve((size_t)ML * XD_N * 4);
  __bf16* u_bf    = (__bf16*)carve((size_t)ML * D_INNER * 2);
  __bf16* y_bf    = (__bf16*)carve((size_t)ML * D_INNER * 2);
  __bf16* hs_bf   = (__bf16*)carve((size_t)ML * D_MODEL * 2);
  __bf16* ipw_bf  = (__bf16*)carve((size_t)XZ_N * D_MODEL * 2);
  __bf16* xpw_bf  = (__bf16*)carve((size_t)XD_N * D_INNER * 2);
  __bf16* dpw_bf  = (__bf16*)carve((size_t)D_INNER * DT_RANK * 2);
  __bf16* opw_bf  = (__bf16*)carve((size_t)D_MODEL * D_INNER * 2);
  __bf16* xdbl_bf = (__bf16*)carve((size_t)ML * XD_N * 2);

  auto cvt = [&](const float* s, __bf16* dst, long n) {
    long nb = (n + 255) / 256;
    f32_to_bf16_kernel<<<dim3((unsigned)nb), dim3(256), 0, stream>>>(s, dst, n);
  };

  // 0) one-time bf16 conversions
  cvt(hs,  hs_bf,  (long)ML * D_MODEL);
  cvt(ipw, ipw_bf, (long)XZ_N * D_MODEL);
  cvt(xpw, xpw_bf, (long)XD_N * D_INNER);
  cvt(dpw, dpw_bf, (long)D_INNER * DT_RANK);
  cvt(opw, opw_bf, (long)D_MODEL * D_INNER);

  // 1) xz = hs @ in_proj_w^T   [8192 x 8192], NT=4 -> 65536 waves (exact)
  {
    int blocks = (ML / 16) * (XZ_N / 64) * 32 / 256;
    gemm_bf16_wmma<4><<<dim3(blocks), dim3(256), 0, stream>>>(
        hs_bf, D_MODEL, ipw_bf, D_MODEL, xz, XZ_N, ML, D_MODEL);
  }

  // 2) conv + SiLU -> u; emit new_conv_state
  {
    long n = (long)ML * D_INNER;
    conv_silu_kernel<<<dim3((unsigned)((n + 255) / 256)), dim3(256), 0, stream>>>(
        xz, cw, cb, u_bf);
    conv_state_kernel<<<dim3((BATCH * D_INNER * D_CONV) / 256), dim3(256), 0, stream>>>(
        xz, out + CS_OFF);
  }

  // 3) x_dbl = u @ x_proj_w^T  [8192 x 160], NT=2 -> 2560 waves (exact)
  {
    int blocks = (ML / 16) * (XD_N / 32) * 32 / 256;
    gemm_bf16_wmma<2><<<dim3(blocks), dim3(256), 0, stream>>>(
        u_bf, D_INNER, xpw_bf, D_INNER, xdbl, XD_N, ML, D_INNER);
  }
  cvt(xdbl, xdbl_bf, (long)ML * XD_N);

  // 4) dt = x_dbl[:, :128] @ dt_proj_w^T -> xz cols [0,4096), NT=4
  {
    int blocks = (ML / 16) * (D_INNER / 64) * 32 / 256;
    gemm_bf16_wmma<4><<<dim3(blocks), dim3(256), 0, stream>>>(
        xdbl_bf, XD_N, dpw_bf, DT_RANK, xz, XZ_N, ML, DT_RANK);
  }

  // 5) selective scan
  scan_kernel<<<dim3((BATCH * D_INNER) / 256), dim3(256), 0, stream>>>(
      xz, u_bf, xdbl, dpb, Amat, Dvec, y_bf, out + SS_OFF);

  // 6) out = y @ out_proj_w^T  [8192 x 2048], NT=4
  {
    int blocks = (ML / 16) * (D_MODEL / 64) * 32 / 256;
    gemm_bf16_wmma<4><<<dim3(blocks), dim3(256), 0, stream>>>(
        y_bf, D_INNER, opw_bf, D_INNER, out, D_MODEL, ML, D_INNER);
  }
}
